// MultiDilateLocalAttention_18588618457406
// MI455X (gfx1250) — matrix operand order repaired and checked by reading, not compiled, not run on hardware
//
#include <hip/hip_runtime.h>
#include <stdint.h>

// ---------------------------------------------------------------------------
// MultiDilateLocalAttention for MI455X (gfx1250, wave32, WMMA).
//   B=8, H=W=64, C=768, 12 heads x 64 dim, dilations (1,2,3), k=3.
//   Pipeline:  qkv GEMM (bf16-split WMMA) -> local attention (+residual)
//              -> proj GEMM (bf16-split WMMA, +bias) -> d_out
//   fp32 accuracy recovered via x = hi(bf16) + lo(bf16); a*b ~ ah*bh+ah*bl+al*bh
// ---------------------------------------------------------------------------

#define NPIX   32768          // B*H*W
#define CDIM   768
#define QKVN   (3 * CDIM)     // 2304
#define NHEAD  12
#define HDIM   64
#define SCALE  0.125f         // 64^-0.5

typedef __attribute__((ext_vector_type(16))) __bf16 v16bf;
typedef __attribute__((ext_vector_type(8)))  __bf16 v8bf;
typedef __attribute__((ext_vector_type(8)))  float  v8f;

// ---- GEMM tiling: block = 256 threads (8 waves), tile 128(M) x 64(N) x 64(K)
//      wave grid 4x2, each wave owns a 32x32 output tile (2x2 WMMA tiles).
#define BM 128
#define BN 64
#define BK 64
#define LSTR (BK + 8)         // 72 shorts: keeps 16B alignment, rotates LDS banks

__device__ __forceinline__ void splitbf(float x, unsigned short& hi, unsigned short& lo)
{
    unsigned xb = __float_as_uint(x);
    unsigned hb = xb & 0xFFFF0000u;          // truncate: hi exact in bf16
    hi = (unsigned short)(hb >> 16);
    float r = x - __uint_as_float(hb);       // exact remainder
    lo = (unsigned short)(__float_as_uint(r) >> 16);
}

__global__ __launch_bounds__(256)
void gemm_bf16x2_wmma(const float* __restrict__ A,    // [M,K] row-major
                      const float* __restrict__ W,    // [N,K] row-major (B^T)
                      const float* __restrict__ bias, // [N] or nullptr
                      float* __restrict__ Cd,         // [M,N] row-major
                      int M, int N, int K)
{
    __shared__ __align__(16) unsigned short As_hi[BM * LSTR];
    __shared__ __align__(16) unsigned short As_lo[BM * LSTR];
    __shared__ __align__(16) unsigned short Bs_hi[BN * LSTR];
    __shared__ __align__(16) unsigned short Bs_lo[BN * LSTR];

    const int tid  = threadIdx.x;
    const int lane = tid & 31;
    const int wv   = tid >> 5;           // wave 0..7
    const int lr   = lane & 15;          // row/col within 16
    const int lhi  = lane >> 4;          // half-wave select
    const int wm   = (wv >> 1) * 32;     // wave M offset in block tile
    const int wn   = (wv & 1) * 32;      // wave N offset in block tile

    const int rowBase = blockIdx.y * BM;
    const int colBase = blockIdx.x * BN;

    v8f acc[2][2] = {};

    for (int kb = 0; kb < K; kb += BK) {
        // ---- stage A panel: BM x BK fp32 -> bf16 hi/lo in LDS -------------
        #pragma unroll
        for (int i = 0; i < (BM * BK) / (4 * 256); ++i) {     // 8 float4/thread
            int idx = tid + i * 256;
            int r   = idx >> 4;                 // 16 float4 per row (BK=64)
            int c4  = (idx & 15) * 4;
            const float4 xa = *(const float4*)(A + (size_t)(rowBase + r) * K + kb + c4);
            unsigned short h0,h1,h2,h3,l0,l1,l2,l3;
            splitbf(xa.x,h0,l0); splitbf(xa.y,h1,l1);
            splitbf(xa.z,h2,l2); splitbf(xa.w,h3,l3);
            uint2 hp, lp;
            hp.x = (unsigned)h0 | ((unsigned)h1 << 16);
            hp.y = (unsigned)h2 | ((unsigned)h3 << 16);
            lp.x = (unsigned)l0 | ((unsigned)l1 << 16);
            lp.y = (unsigned)l2 | ((unsigned)l3 << 16);
            *(uint2*)&As_hi[r * LSTR + c4] = hp;
            *(uint2*)&As_lo[r * LSTR + c4] = lp;
        }
        // ---- stage B panel: BN x BK fp32 -> bf16 hi/lo in LDS -------------
        #pragma unroll
        for (int i = 0; i < (BN * BK) / (4 * 256); ++i) {     // 4 float4/thread
            int idx = tid + i * 256;
            int r   = idx >> 4;
            int c4  = (idx & 15) * 4;
            const float4 xb = *(const float4*)(W + (size_t)(colBase + r) * K + kb + c4);
            unsigned short h0,h1,h2,h3,l0,l1,l2,l3;
            splitbf(xb.x,h0,l0); splitbf(xb.y,h1,l1);
            splitbf(xb.z,h2,l2); splitbf(xb.w,h3,l3);
            uint2 hp, lp;
            hp.x = (unsigned)h0 | ((unsigned)h1 << 16);
            hp.y = (unsigned)h2 | ((unsigned)h3 << 16);
            lp.x = (unsigned)l0 | ((unsigned)l1 << 16);
            lp.y = (unsigned)l2 | ((unsigned)l3 << 16);
            *(uint2*)&Bs_hi[r * LSTR + c4] = hp;
            *(uint2*)&Bs_lo[r * LSTR + c4] = lp;
        }
        // prefetch next K-panel (emits global_prefetch_b8)
        if (kb + BK < K) {
            __builtin_prefetch(A + (size_t)(rowBase + (tid >> 1)) * K + (kb + BK) + (tid & 1) * 32, 0, 1);
            if (tid < 128)
                __builtin_prefetch(W + (size_t)(colBase + (tid >> 1)) * K + (kb + BK) + (tid & 1) * 32, 0, 1);
        }
        __syncthreads();

        // ---- compute: 2 k-steps of 32, 2x2 tiles, 3 WMMA each (hi/lo split)
        #pragma unroll
        for (int ks = 0; ks < BK; ks += 32) {
            union Frag { v16bf v; v8bf h[2]; };
            Frag ah[2], al[2], bh[2], bl[2];
            // A fragment (16-bit A layout, ISA 7.12.2):
            //   lanes 0-15:  v0..3 = K0..7,  v4..7 = K16..23
            //   lanes 16-31: v0..3 = K8..15, v4..7 = K24..31
            #pragma unroll
            for (int mt = 0; mt < 2; ++mt) {
                const unsigned short* ap = &As_hi[(wm + mt * 16 + lr) * LSTR + ks + lhi * 8];
                ah[mt].h[0] = *(const v8bf*)ap;
                ah[mt].h[1] = *(const v8bf*)(ap + 16);
                const unsigned short* alp = &As_lo[(wm + mt * 16 + lr) * LSTR + ks + lhi * 8];
                al[mt].h[0] = *(const v8bf*)alp;
                al[mt].h[1] = *(const v8bf*)(alp + 16);
            }
            // B fragment (same K order as A): lanes 0-15 hold K ks+0..7,16..23 of column lr,
            //             lanes 16-31 hold K ks+8..15,24..31
            #pragma unroll
            for (int nt = 0; nt < 2; ++nt) {
                const unsigned short* bp = &Bs_hi[(wn + nt * 16 + lr) * LSTR + ks + lhi * 8];
                bh[nt].h[0] = *(const v8bf*)bp;
                bh[nt].h[1] = *(const v8bf*)(bp + 16);
                const unsigned short* blp = &Bs_lo[(wn + nt * 16 + lr) * LSTR + ks + lhi * 8];
                bl[nt].h[0] = *(const v8bf*)blp;
                bl[nt].h[1] = *(const v8bf*)(blp + 16);
            }
            #pragma unroll
            for (int mt = 0; mt < 2; ++mt)
                #pragma unroll
                for (int nt = 0; nt < 2; ++nt) {
                    acc[mt][nt] = __builtin_amdgcn_wmma_f32_16x16x32_bf16(
                        false, ah[mt].v, false, bh[nt].v, (short)0, acc[mt][nt], false, false);
                    acc[mt][nt] = __builtin_amdgcn_wmma_f32_16x16x32_bf16(
                        false, ah[mt].v, false, bl[nt].v, (short)0, acc[mt][nt], false, false);
                    acc[mt][nt] = __builtin_amdgcn_wmma_f32_16x16x32_bf16(
                        false, al[mt].v, false, bh[nt].v, (short)0, acc[mt][nt], false, false);
                }
        }
        __syncthreads();
    }

    // ---- epilogue: C/D layout (VGPR j, lanes<16 -> M=j, lanes>=16 -> M=8+j)
    #pragma unroll
    for (int mt = 0; mt < 2; ++mt)
        #pragma unroll
        for (int nt = 0; nt < 2; ++nt) {
            const int col = colBase + wn + nt * 16 + lr;
            const float bsum = bias ? bias[col] : 0.0f;
            #pragma unroll
            for (int j = 0; j < 8; ++j) {
                const int row = rowBase + wm + mt * 16 + lhi * 8 + j;
                Cd[(size_t)row * N + col] = acc[mt][nt][j] + bsum;
            }
        }
}

// ---------------------------------------------------------------------------
// Local attention: one thread per (pixel, head).  qkv row = [q|k|v], each 768,
// channel c of head h lives at h*64+c (dilation-major == head-major here).
// Zero-padding semantics: OOB neighbor has k=0 -> logit 0 participates in
// softmax; v=0 -> no numerator contribution.  Residual +x fused into y0.
// ---------------------------------------------------------------------------
__global__ __launch_bounds__(256)
void mdla_attn(const float* __restrict__ qkv,  // [NPIX, 2304]
               const float* __restrict__ x,    // [NPIX, 768]
               float* __restrict__ y0)         // [NPIX, 768]
{
    const int t    = blockIdx.x * blockDim.x + threadIdx.x;
    const int head = t >> 15;                  // 0..11
    const int p    = t & (NPIX - 1);
    if (head >= NHEAD) return;

    const int px = p & 63;
    const int py = (p >> 6) & 63;
    const int b  = p >> 12;
    const int dil = (head >> 2) + 1;           // 4 heads per dilation 1,2,3

    const float4* qr = (const float4*)(qkv + (size_t)p * QKVN + head * HDIM);
    float4 q4[16];
    #pragma unroll
    for (int c = 0; c < 16; ++c) q4[c] = qr[c];

    float lg[9];
    int   npix[9];
    unsigned okm = 0;
    #pragma unroll
    for (int j = 0; j < 9; ++j) {
        const int ny = py + (j / 3 - 1) * dil;
        const int nx = px + (j % 3 - 1) * dil;
        const bool ok = ((unsigned)ny < 64u) && ((unsigned)nx < 64u);
        npix[j] = (b * 64 + ny) * 64 + nx;     // only used when ok
        float dot = 0.0f;
        if (ok) {
            okm |= 1u << j;
            const float4* kr = (const float4*)(qkv + (size_t)npix[j] * QKVN + CDIM + head * HDIM);
            #pragma unroll
            for (int c = 0; c < 16; ++c) {
                const float4 kv = kr[c];
                dot = fmaf(q4[c].x, kv.x, dot);
                dot = fmaf(q4[c].y, kv.y, dot);
                dot = fmaf(q4[c].z, kv.z, dot);
                dot = fmaf(q4[c].w, kv.w, dot);
            }
        }
        lg[j] = dot * SCALE;                   // OOB -> 0, matches zero-pad ref
    }

    float mx = lg[0];
    #pragma unroll
    for (int j = 1; j < 9; ++j) mx = fmaxf(mx, lg[j]);
    float wt[9], den = 0.0f;
    #pragma unroll
    for (int j = 0; j < 9; ++j) { wt[j] = __expf(lg[j] - mx); den += wt[j]; }
    const float inv = 1.0f / den;

    float4 o4[16] = {};
    #pragma unroll
    for (int j = 0; j < 9; ++j) {
        if (okm & (1u << j)) {
            const float w = wt[j];
            const float4* vr = (const float4*)(qkv + (size_t)npix[j] * QKVN + 2 * CDIM + head * HDIM);
            #pragma unroll
            for (int c = 0; c < 16; ++c) {
                const float4 vv = vr[c];
                o4[c].x = fmaf(w, vv.x, o4[c].x);
                o4[c].y = fmaf(w, vv.y, o4[c].y);
                o4[c].z = fmaf(w, vv.z, o4[c].z);
                o4[c].w = fmaf(w, vv.w, o4[c].w);
            }
        }
    }

    const float4* xr = (const float4*)(x  + (size_t)p * CDIM + head * HDIM);
    float4*       yr = (float4*)      (y0 + (size_t)p * CDIM + head * HDIM);
    #pragma unroll
    for (int c = 0; c < 16; ++c) {
        const float4 xv = xr[c];
        float4 ov;
        ov.x = o4[c].x * inv + xv.x;
        ov.y = o4[c].y * inv + xv.y;
        ov.z = o4[c].z * inv + xv.z;
        ov.w = o4[c].w * inv + xv.w;
        yr[c] = ov;
    }
}

// ---------------------------------------------------------------------------
extern "C" void kernel_launch(void* const* d_in, const int* in_sizes, int n_in,
                              void* d_out, int out_size, void* d_ws, size_t ws_size,
                              hipStream_t stream)
{
    const float* x      = (const float*)d_in[0];   // [8,64,64,768]
    const float* qkv_w  = (const float*)d_in[1];   // [2304,768]
    const float* proj_w = (const float*)d_in[2];   // [768,768]
    const float* proj_b = (const float*)d_in[3];   // [768]
    float* out = (float*)d_out;                    // [8,64,64,768]

    // workspace: qkv (32768*2304 f32 = 288MB) + y0 (32768*768 f32 = 96MB)
    float* qkv = (float*)d_ws;
    float* y0  = qkv + (size_t)NPIX * QKVN;

    dim3 blk(256);
    gemm_bf16x2_wmma<<<dim3(QKVN / BN, NPIX / BM), blk, 0, stream>>>(
        x, qkv_w, nullptr, qkv, NPIX, QKVN, CDIM);
    mdla_attn<<<dim3((NPIX * NHEAD) / 256), blk, 0, stream>>>(qkv, x, y0);
    gemm_bf16x2_wmma<<<dim3(CDIM / BN, NPIX / BM), blk, 0, stream>>>(
        y0, proj_w, proj_b, out, NPIX, CDIM, CDIM);
}
